// VectorQuantizer_35029753266884
// MI455X (gfx1250) — compile-verified, matrix operand order checked by reading, not run on hardware
//
#include <hip/hip_runtime.h>
#include <hip/hip_bf16.h>
#include <cstdint>

#define NUM_CODES 4096
#define CODE_DIM  256
#define NTOK      32768   // 32 * 1024
#define BETA      0.25f
#define EPSV      1e-8f

#define TILE_CODES 32                      // codes staged per TDM tile
#define NTILES     (NUM_CODES / TILE_CODES)

typedef __bf16 v16bf __attribute__((ext_vector_type(16)));
typedef float  v8f   __attribute__((ext_vector_type(8)));
typedef unsigned int u32x4 __attribute__((ext_vector_type(4)));
typedef int          i32x4 __attribute__((ext_vector_type(4)));
typedef int          i32x8 __attribute__((ext_vector_type(8)));

#if __has_builtin(__builtin_amdgcn_tensor_load_to_lds) && __has_builtin(__builtin_amdgcn_s_wait_tensorcnt)
#define HAVE_TDM 1
#else
#define HAVE_TDM 0
#endif

union BF16Frag {
    unsigned short us[16];
    uint4 q[2];
    v16bf v;
};

__device__ __forceinline__ void split_bf16(float x, unsigned short& hi, unsigned short& lo) {
    unsigned u = __float_as_uint(x);
    unsigned short h = (unsigned short)(u >> 16);     // truncate -> exact residual
    float hf = __uint_as_float(((unsigned)h) << 16);
    float r  = x - hf;
    hi = h;
    lo = (unsigned short)(__float_as_uint(r) >> 16);
}

// ---------------------------------------------------------------------------
// Kernel 0: zero the atomic accumulators (fresh every launch, graph-safe)
// ---------------------------------------------------------------------------
__global__ void vq_init(unsigned int* __restrict__ counts,
                        float* __restrict__ loss_sum,
                        unsigned int* __restrict__ mask_sum) {
    int i = blockIdx.x * blockDim.x + threadIdx.x;
    if (i < NUM_CODES) counts[i] = 0u;
    if (i == 0) { *loss_sum = 0.0f; *mask_sum = 0u; }
}

// ---------------------------------------------------------------------------
// Kernel 1: codebook -> bf16 hi/lo split + ||e||^2 per code
// ---------------------------------------------------------------------------
__global__ __launch_bounds__(256) void vq_prep(const float* __restrict__ e_w,
                                               unsigned short* __restrict__ e_hi,
                                               unsigned short* __restrict__ e_lo,
                                               float* __restrict__ e_norm) {
    int r = blockIdx.x;
    int t = threadIdx.x;
    float x = e_w[(size_t)r * CODE_DIM + t];
    unsigned short h, l;
    split_bf16(x, h, l);
    e_hi[(size_t)r * CODE_DIM + t] = h;
    e_lo[(size_t)r * CODE_DIM + t] = l;
    __shared__ float red[256];
    red[t] = x * x;
    __syncthreads();
    for (int s = 128; s > 0; s >>= 1) {
        if (t < s) red[t] += red[t + s];
        __syncthreads();
    }
    if (t == 0) e_norm[r] = red[0];
}

#if HAVE_TDM
// Issue one TDM 2D tile load: TILE_CODES rows x 256 bf16 -> LDS.
// D# per CDNA5 ISA 8.3/8.4: group0 {count, lds_addr, global_addr, type=2},
// group1 {mask=0, data_size=2B, tensor dims/strides, tile dims}, groups2/3 zero (2D).
// clang-23 6-arg form: (u32x4 g0, i32x8 g1, i32x4, i32x4, i32x8, i32 cpol)
__device__ __forceinline__ void tdm_load_tile(const unsigned short* gsrc, unsigned lds_off) {
    unsigned long long ga = (unsigned long long)(uintptr_t)gsrc;
    u32x4 g0;
    g0[0] = 1u;                                                // count=1, user descriptor
    g0[1] = lds_off;                                           // lds_addr (bytes)
    g0[2] = (unsigned)(ga & 0xffffffffu);                      // global_addr[31:0]
    g0[3] = (unsigned)((ga >> 32) & 0x01ffffffu) | 0x80000000u;// global_addr[56:32] | type=2
    i32x8 g1;
    g1[0] = 0x00010000;            // workgroup_mask=0, data_size=1 (2 bytes)
    g1[1] = (int)(256u << 16);     // tensor_dim0 = 256     (bits 79:48 low half)
    g1[2] = (int)(4096u << 16);    // tensor_dim1 = 4096    (bits 111:80 low half)
    g1[3] = (int)(256u << 16);     // tile_dim0 = 256       (bits 127:112)
    g1[4] = TILE_CODES;            // tile_dim1             (bits 143:128)
    g1[5] = 256;                   // tensor_dim0_stride = 256 (bits 207:160 low)
    g1[6] = 0;
    g1[7] = 0;
    i32x4 z4 = {0, 0, 0, 0};
    i32x8 z8 = {0, 0, 0, 0, 0, 0, 0, 0};
    __builtin_amdgcn_tensor_load_to_lds(g0, g1, z4, z4, z8, 0);
}
#endif

// ---------------------------------------------------------------------------
// Kernel 2: fused distance GEMM (WMMA bf16 hi/lo split) + row argmin.
// 256 threads = 8 waves, each wave owns 16 z-rows. Codebook tiles staged
// into LDS by the Tensor Data Mover (double-buffered), shared by all waves.
// ---------------------------------------------------------------------------
__global__ __launch_bounds__(256) void vq_argmin(
    const float* __restrict__ z_e,
    const unsigned short* __restrict__ e_hi,
    const unsigned short* __restrict__ e_lo,
    const float* __restrict__ e_norm,
    int* __restrict__ out_idx)
{
    const int lane    = threadIdx.x & 31;
    const int wave    = threadIdx.x >> 5;
    const int rowBase = blockIdx.x * 128 + wave * 16;
    const int myRow   = rowBase + (lane & 15);
    const int koff    = (lane & 16) ? 8 : 0;   // ISA 16-bit A layout: lanes 16-31 take K+8

    // ---- preload full-K A fragments (hi/lo) into registers: 8 K-steps ----
    BF16Frag aHi[8], aLo[8];
    const float* zr = z_e + (size_t)myRow * CODE_DIM;
#pragma unroll
    for (int ks = 0; ks < 8; ++ks) {
        const int k0 = ks * 32 + koff;
        float4 c0a = *(const float4*)(zr + k0);
        float4 c0b = *(const float4*)(zr + k0 + 4);
        float4 c1a = *(const float4*)(zr + k0 + 16);
        float4 c1b = *(const float4*)(zr + k0 + 20);
        float v0[8] = {c0a.x, c0a.y, c0a.z, c0a.w, c0b.x, c0b.y, c0b.z, c0b.w};
        float v1[8] = {c1a.x, c1a.y, c1a.z, c1a.w, c1b.x, c1b.y, c1b.z, c1b.w};
#pragma unroll
        for (int j = 0; j < 8; ++j) {
            split_bf16(v0[j], aHi[ks].us[j],     aLo[ks].us[j]);
            split_bf16(v1[j], aHi[ks].us[8 + j], aLo[ks].us[8 + j]);
        }
    }

    // per-lane running argmin: VGPR i of C/D frag <-> row i (lanes 0-15) / i+8 (16-31)
    float minv[8];
    int   mini[8];
#pragma unroll
    for (int i = 0; i < 8; ++i) { minv[i] = 3.4e38f; mini[i] = 0x7fffffff; }

#if HAVE_TDM
    // double-buffered LDS staging of hi/lo codebook tiles (TDM, wave 0 only)
    __shared__ __align__(16) unsigned short ldsHi[2][TILE_CODES * CODE_DIM];
    __shared__ __align__(16) unsigned short ldsLo[2][TILE_CODES * CODE_DIM];

    if (wave == 0) {
        tdm_load_tile(e_hi, (unsigned)(uintptr_t)&ldsHi[0][0]);
        tdm_load_tile(e_lo, (unsigned)(uintptr_t)&ldsLo[0][0]);
    }

    for (int t = 0; t < NTILES; ++t) {
        const int cur = t & 1;
        if (wave == 0) {
            if (t + 1 < NTILES) {
                const size_t goff = (size_t)(t + 1) * TILE_CODES * CODE_DIM;
                tdm_load_tile(e_hi + goff, (unsigned)(uintptr_t)&ldsHi[cur ^ 1][0]);
                tdm_load_tile(e_lo + goff, (unsigned)(uintptr_t)&ldsLo[cur ^ 1][0]);
                __builtin_amdgcn_s_wait_tensorcnt(2);   // tile t's 2 ops complete (in-order)
            } else {
                __builtin_amdgcn_s_wait_tensorcnt(0);
            }
        }
        __syncthreads();                                // tile t visible to all waves

        const int cb = t * TILE_CODES;
#pragma unroll
        for (int s = 0; s < TILE_CODES / 16; ++s) {
            const int code = cb + s * 16 + (lane & 15); // B column for this lane
            const unsigned short* bh = &ldsHi[cur][(s * 16 + (lane & 15)) * CODE_DIM];
            const unsigned short* bl = &ldsLo[cur][(s * 16 + (lane & 15)) * CODE_DIM];

            v8f acc = {0.f, 0.f, 0.f, 0.f, 0.f, 0.f, 0.f, 0.f};
#pragma unroll
            for (int ks = 0; ks < 8; ++ks) {
                const int k0 = ks * 32 + koff;
                BF16Frag bHi, bLo;
                bHi.q[0] = *(const uint4*)(bh + k0);
                bHi.q[1] = *(const uint4*)(bh + k0 + 16);
                bLo.q[0] = *(const uint4*)(bl + k0);
                bLo.q[1] = *(const uint4*)(bl + k0 + 16);
                acc = __builtin_amdgcn_wmma_f32_16x16x32_bf16(false, aHi[ks].v, false, bHi.v,
                                                              (short)0, acc, false, false);
                acc = __builtin_amdgcn_wmma_f32_16x16x32_bf16(false, aHi[ks].v, false, bLo.v,
                                                              (short)0, acc, false, false);
                acc = __builtin_amdgcn_wmma_f32_16x16x32_bf16(false, aLo[ks].v, false, bHi.v,
                                                              (short)0, acc, false, false);
            }
            const float en = e_norm[code];
#pragma unroll
            for (int i = 0; i < 8; ++i) {
                const float d = en - 2.0f * acc[i];     // ||z||^2 is row-constant: drop
                if (d < minv[i] || (d == minv[i] && code < mini[i])) {
                    minv[i] = d; mini[i] = code;
                }
            }
        }
        __syncthreads();                                // all done reading buffer `cur`
    }
#else
    // fallback: direct global loads of the codebook
    for (int cb = 0; cb < NUM_CODES; cb += 64) {
#pragma unroll
        for (int s = 0; s < 4; ++s) {
            const int code = cb + s * 16 + (lane & 15);
            const unsigned short* bh = e_hi + (size_t)code * CODE_DIM;
            const unsigned short* bl = e_lo + (size_t)code * CODE_DIM;
            if (code + 64 < NUM_CODES)
                __builtin_prefetch(e_hi + (size_t)(code + 64) * CODE_DIM, 0, 1);
            v8f acc = {0.f, 0.f, 0.f, 0.f, 0.f, 0.f, 0.f, 0.f};
#pragma unroll
            for (int ks = 0; ks < 8; ++ks) {
                const int k0 = ks * 32 + koff;
                BF16Frag bHi, bLo;
                bHi.q[0] = *(const uint4*)(bh + k0);
                bHi.q[1] = *(const uint4*)(bh + k0 + 16);
                bLo.q[0] = *(const uint4*)(bl + k0);
                bLo.q[1] = *(const uint4*)(bl + k0 + 16);
                acc = __builtin_amdgcn_wmma_f32_16x16x32_bf16(false, aHi[ks].v, false, bHi.v,
                                                              (short)0, acc, false, false);
                acc = __builtin_amdgcn_wmma_f32_16x16x32_bf16(false, aHi[ks].v, false, bLo.v,
                                                              (short)0, acc, false, false);
                acc = __builtin_amdgcn_wmma_f32_16x16x32_bf16(false, aLo[ks].v, false, bHi.v,
                                                              (short)0, acc, false, false);
            }
            const float en = e_norm[code];
#pragma unroll
            for (int i = 0; i < 8; ++i) {
                const float d = en - 2.0f * acc[i];
                if (d < minv[i] || (d == minv[i] && code < mini[i])) {
                    minv[i] = d; mini[i] = code;
                }
            }
        }
    }
#endif

    // cross-lane argmin within each 16-lane half (first-index tie-break)
#pragma unroll
    for (int m = 1; m < 16; m <<= 1) {
#pragma unroll
        for (int i = 0; i < 8; ++i) {
            float ov = __shfl_xor(minv[i], m, 32);
            int   oi = __shfl_xor(mini[i], m, 32);
            if (ov < minv[i] || (ov == minv[i] && oi < mini[i])) {
                minv[i] = ov; mini[i] = oi;
            }
        }
    }
    if (lane == 0) {
#pragma unroll
        for (int i = 0; i < 8; ++i) out_idx[rowBase + i] = mini[i];
    } else if (lane == 16) {
#pragma unroll
        for (int i = 0; i < 8; ++i) out_idx[rowBase + 8 + i] = mini[i];
    }
}

// ---------------------------------------------------------------------------
// Kernel 3: gather z_q (== z_q_st in forward), losses, histogram
// ---------------------------------------------------------------------------
__global__ __launch_bounds__(256) void vq_gather(
    const float* __restrict__ z_e, const int* __restrict__ mask,
    const float* __restrict__ e_w, const int* __restrict__ idx,
    float* __restrict__ out,
    float* __restrict__ loss_sum,
    unsigned int* __restrict__ counts,
    unsigned int* __restrict__ mask_sum)
{
    const int n = blockIdx.x;
    const int t = threadIdx.x;
    const int code = idx[n];
    const int m    = mask[n];

    const float ze = z_e[(size_t)n * CODE_DIM + t];
    const float eq = e_w[(size_t)code * CODE_DIM + t];
    const float zq = m ? eq : 0.0f;

    out[(size_t)n * CODE_DIM + t] = zq;                                   // z_q
    out[(size_t)NTOK * CODE_DIM + (size_t)n * CODE_DIM + t] = zq;         // z_q_st (same fwd)

    const float diff = zq - ze;
    __shared__ float red[256];
    red[t] = m ? diff * diff : 0.0f;
    __syncthreads();
    for (int s = 128; s > 0; s >>= 1) {
        if (t < s) red[t] += red[t + s];
        __syncthreads();
    }
    if (t == 0) {
        atomicAdd(loss_sum, red[0]);
        out[(size_t)2 * NTOK * CODE_DIM + n] = (float)code;               // indices
        if (m) atomicAdd(&counts[code], 1u);
        atomicAdd(mask_sum, (unsigned)m);
    }
}

// ---------------------------------------------------------------------------
// Kernel 4: scalars (vq_loss, codebook, commitment, perplexity)
// ---------------------------------------------------------------------------
__global__ __launch_bounds__(256) void vq_final(const unsigned int* __restrict__ counts,
                                                const float* __restrict__ loss_sum,
                                                const unsigned int* __restrict__ mask_sum,
                                                float* __restrict__ out) {
    const int t = threadIdx.x;
    const float tot = (float)(*mask_sum);
    const float denom = tot + EPSV;
    float ent = 0.0f;
    for (int i = t; i < NUM_CODES; i += 256) {
        const float p = (float)counts[i] / denom;
        ent -= p * logf(p + EPSV);
    }
    __shared__ float red[256];
    red[t] = ent;
    __syncthreads();
    for (int s = 128; s > 0; s >>= 1) {
        if (t < s) red[t] += red[t + s];
        __syncthreads();
    }
    if (t == 0) {
        const float valid = fmaxf(tot, 1.0f);
        const float loss  = *loss_sum / valid;   // codebook == commitment in forward
        const size_t base = (size_t)2 * NTOK * CODE_DIM + NTOK;
        out[base + 0] = loss * (1.0f + BETA);    // vq_loss
        out[base + 1] = loss;                    // codebook_loss
        out[base + 2] = loss;                    // commitment_loss
        out[base + 3] = expf(red[0]);            // perplexity
    }
}

// ---------------------------------------------------------------------------
extern "C" void kernel_launch(void* const* d_in, const int* in_sizes, int n_in,
                              void* d_out, int out_size, void* d_ws, size_t ws_size,
                              hipStream_t stream) {
    (void)in_sizes; (void)n_in; (void)out_size; (void)ws_size;
    const float* z_e    = (const float*)d_in[0];
    const int*   mask   = (const int*)d_in[1];
    const float* e_w    = (const float*)d_in[2];
    float*       out    = (float*)d_out;

    // workspace layout
    char* ws = (char*)d_ws;
    unsigned short* e_hi   = (unsigned short*)(ws);                         // 2 MB
    unsigned short* e_lo   = (unsigned short*)(ws + 2097152);               // 2 MB
    float*          e_norm = (float*)(ws + 4194304);                        // 16 KB
    int*            idx    = (int*)(ws + 4210688);                          // 128 KB
    unsigned int*   counts = (unsigned int*)(ws + 4341760);                 // 16 KB
    float*          lsum   = (float*)(ws + 4358144);
    unsigned int*   msum   = (unsigned int*)(ws + 4358148);

    vq_init  <<<(NUM_CODES + 255) / 256, 256, 0, stream>>>(counts, lsum, msum);
    vq_prep  <<<NUM_CODES, 256, 0, stream>>>(e_w, e_hi, e_lo, e_norm);
    vq_argmin<<<NTOK / 128, 256, 0, stream>>>(z_e, e_hi, e_lo, e_norm, idx);
    vq_gather<<<NTOK, 256, 0, stream>>>(z_e, mask, e_w, idx, out, lsum, counts, msum);
    vq_final <<<1, 256, 0, stream>>>(counts, lsum, msum, out);
}
